// TransformerXLEncoder_16140487098860
// MI455X (gfx1250) — matrix operand order repaired and checked by reading, not compiled, not run on hardware
//
#include <hip/hip_runtime.h>
#include <stdint.h>

// ---------------------------------------------------------------------------
// Transformer-XL encoder for MI455X (gfx1250): all matmuls via
// v_wmma_f32_16x16x32_bf16 (wave32 WMMA), f32 accumulation.
// L=1024, B=2, D=1024, NH=16, DH=64, NL=4, MLEN=1024, KLEN=2048.
// ---------------------------------------------------------------------------

typedef __attribute__((ext_vector_type(16))) __bf16 v16bf;
typedef __attribute__((ext_vector_type(8)))  float  v8f;

union BF16Frag {
    v16bf    v;
    __bf16   e[16];
    unsigned u[8];
};

// native conversions (compiler emits hardware cvt ops on gfx1250)
static __device__ __forceinline__ __bf16 f2bf(float f) { return (__bf16)f; }
static __device__ __forceinline__ float  bf2f(__bf16 b) { return (float)b; }

static __device__ __forceinline__ v8f wmma_bf16(const BF16Frag& a, const BF16Frag& b, v8f c) {
    // D(16x16,f32) = A(16x32,bf16) * B(32x16,bf16) + C
    return __builtin_amdgcn_wmma_f32_16x16x32_bf16(false, a.v, false, b.v, (short)0, c, false, false);
}

// async global->LDS staging (CDNA5 ASYNCcnt path), guarded for toolchain safety
#if __has_builtin(__builtin_amdgcn_global_load_async_to_lds_b128) && \
    __has_builtin(__builtin_amdgcn_s_wait_asynccnt)
#define USE_ASYNC_LDS 1
// builtin signature (from compiler diagnostic): (int4 AS1*, int4 AS3*, imm, imm)
typedef int v4i_vs __attribute__((vector_size(4 * sizeof(int))));
typedef __attribute__((address_space(1))) v4i_vs* async_gptr;
typedef __attribute__((address_space(3))) v4i_vs* async_lptr;
#else
#define USE_ASYNC_LDS 0
#endif

// dims
#define DIM_L    1024
#define DIM_B    2
#define DIM_D    1024
#define DIM_NH   16
#define DIM_DH   64
#define DIM_NL   4
#define DIM_MLEN 1024
#define DIM_KLEN 2048
#define ROWS_Q   2048      /* L*B          */
#define ROWS_CAT 4096      /* KLEN*B       */
#define QKV_N    3072      /* 3*NH*DH      */
#define CORE_ELEMS (2097152ll)  /* L*B*D */

// ---------------------------------------------------------------------------
// elementwise helpers
// ---------------------------------------------------------------------------
__global__ void k_f32_to_bf16(const float* __restrict__ in, __bf16* __restrict__ out, long n) {
    long i = (long)blockIdx.x * blockDim.x + threadIdx.x;
    if (i < n) out[i] = f2bf(in[i]);
}
__global__ void k_copy_f32(const float* __restrict__ in, float* __restrict__ out, long n) {
    long i = (long)blockIdx.x * blockDim.x + threadIdx.x;
    if (i < n) out[i] = in[i];
}
__global__ void k_concat_bf16(const float* __restrict__ memL, const float* __restrict__ core,
                              __bf16* __restrict__ cat, long n) {
    long i = (long)blockIdx.x * blockDim.x + threadIdx.x;
    if (i >= n) return;
    float v = (i < CORE_ELEMS) ? memL[i] : core[i - CORE_ELEMS];
    cat[i] = f2bf(v);
}
// pos_emb[p, c], p in [0,KLEN): pos = KLEN-1-p; c<512 -> sin, else cos
__global__ void k_pos_emb(__bf16* __restrict__ pe) {
    long i = (long)blockIdx.x * blockDim.x + threadIdx.x;
    if (i >= (long)DIM_KLEN * DIM_D) return;
    int p = (int)(i >> 10), c = (int)(i & 1023);
    float pos = (float)(DIM_KLEN - 1 - p);
    int kf = (c < 512) ? c : c - 512;
    float inv = __expf(-((float)(2 * kf) * (1.0f / 1024.0f)) * 9.210340371976184f); // ln(1e4)
    float ang = pos * inv;
    pe[i] = f2bf((c < 512) ? __sinf(ang) : __cosf(ang));
}

// ---------------------------------------------------------------------------
// bf16 WMMA GEMM: C[M,N] = A[M,K] @ B[K,N], A/B bf16 row-major.
// 128x128 block tile, 8 waves (256 thr); wave grid 4(M) x 2(N); each wave
// owns a 32x64 sub-tile = 2x4 WMMA(16x16x32) accumulators.
// M % 128 == 0, N % 128 == 0, K % 32 == 0 (true for all shapes here).
// EPI: 0 = store f32, 1 = store bf16, 2 = +bias, relu, store bf16,
//      3 = +bias, store f32
// ---------------------------------------------------------------------------
template <int EPI>
__global__ __launch_bounds__(256) void k_gemm(const __bf16* __restrict__ A,
                                              const __bf16* __restrict__ B,
                                              void* __restrict__ Cout,
                                              const float* __restrict__ bias,
                                              int N, int K)
{
    __shared__ __bf16 As[128][32];   // [m][k]
    __shared__ __bf16 Bs[128][32];   // [n][k]  (transposed so K is contiguous)
    const int m0 = blockIdx.y * 128;
    const int n0 = blockIdx.x * 128;
    const int tid  = threadIdx.x;
    const int lane = tid & 31;
    const int wave = tid >> 5;
    const int wm = (wave >> 1) * 32;   // 4 wave-rows
    const int wn = (wave & 1) * 64;    // 2 wave-cols
    const int ml = lane & 15;
    const int hf = lane >> 4;

    v8f acc[2][4] = {};

    for (int kb = 0; kb < K; kb += 32) {
        __syncthreads();
        // ---- A tile: 128 x 32 bf16 (rows contiguous, 64B each)
#if USE_ASYNC_LDS
        for (int t = tid; t < 128 * 4; t += 256) {          // 512 x 16B segments
            int r = t >> 2, c = (t & 3) * 8;                // 8 bf16 per segment
            __builtin_amdgcn_global_load_async_to_lds_b128(
                (async_gptr)(A + (size_t)(m0 + r) * K + kb + c),
                (async_lptr)&As[r][c], 0, 0);
        }
#else
        for (int t = tid; t < 128 * 8; t += 256) {          // 1024 x 8B
            int r = t >> 3, c = (t & 7) * 4;
            *(unsigned long long*)&As[r][c] =
                *(const unsigned long long*)(A + (size_t)(m0 + r) * K + kb + c);
        }
#endif
        // ---- B tile transposed into [n][k] (strided gather)
        for (int t = tid; t < 128 * 32; t += 256) {
            int n = t >> 5, k = t & 31;
            Bs[n][k] = B[(size_t)(kb + k) * N + n0 + n];
        }
#if USE_ASYNC_LDS
        __builtin_amdgcn_s_wait_asynccnt(0);
#endif
        __syncthreads();

        BF16Frag af[2], bfr[4];
        #pragma unroll
        for (int tm = 0; tm < 2; ++tm) {
            int m = wm + tm * 16 + ml;
            #pragma unroll
            for (int p = 0; p < 8; ++p) {
                int k = 2 * (p & 3) + 16 * (p >> 2) + 8 * hf;   // CDNA5 16-bit A layout
                af[tm].u[p] = *(const unsigned*)&As[m][k];
            }
        }
        #pragma unroll
        for (int tn = 0; tn < 4; ++tn) {
            int n = wn + tn * 16 + ml;
            #pragma unroll
            for (int p = 0; p < 8; ++p) {
                int k = 2 * (p & 3) + 16 * (p >> 2) + 8 * hf;   // CDNA5 16-bit A layout (matches A)
                bfr[tn].u[p] = *(const unsigned*)&Bs[n][k];
            }
        }
        #pragma unroll
        for (int tm = 0; tm < 2; ++tm)
            #pragma unroll
            for (int tn = 0; tn < 4; ++tn)
                acc[tm][tn] = wmma_bf16(af[tm], bfr[tn], acc[tm][tn]);
    }

    // epilogue: D layout -> element (M = v + 8*half, N = lane&15)
    #pragma unroll
    for (int tm = 0; tm < 2; ++tm)
        #pragma unroll
        for (int tn = 0; tn < 4; ++tn)
            #pragma unroll
            for (int v = 0; v < 8; ++v) {
                int row = m0 + wm + tm * 16 + v + 8 * hf;
                int col = n0 + wn + tn * 16 + ml;
                float x = acc[tm][tn][v];
                size_t idx = (size_t)row * N + col;
                if (EPI == 0)      ((float*)Cout)[idx] = x;
                else if (EPI == 1) ((__bf16*)Cout)[idx] = f2bf(x);
                else if (EPI == 2) { x += bias[col]; x = x > 0.f ? x : 0.f; ((__bf16*)Cout)[idx] = f2bf(x); }
                else               { x += bias[col]; ((float*)Cout)[idx] = x; }
            }
}

// ---------------------------------------------------------------------------
// Relative-position attention, one (batch, head, 16-query-row tile) per WG.
// Full 16x2048 score strip in LDS (128 KB of the 320 KB WGP LDS).
//   AC[i,j]  = (q_i + r_w_bias_h) . k_j          (WMMA)
//   BDraw    = (q_i + r_r_bias_h) . rk_jj        (WMMA), shifted: j = jj-1023+i
//   masked softmax (valid j <= 1024+i), then out = prob @ V  (WMMA)
// Causal clipping: AC tiles j0 < i0+1040; BD tiles jj0 >= 1008-i0;
// PV K-loop kb < i0+1040 (rounded to 32). Wrap region of the shift is exactly
// the masked region, so it is simply skipped.
// wh = w_heads bf16 [4096, 3072] rows=(j*2+b); rk bf16 [2048, 1024].
// av out bf16 [2048, 1024] rows=(i*2+b).
// ---------------------------------------------------------------------------
__global__ __launch_bounds__(128) void k_attention(const __bf16* __restrict__ wh,
                                                   const __bf16* __restrict__ rk,
                                                   const float* __restrict__ rwb,
                                                   const float* __restrict__ rrb,
                                                   __bf16* __restrict__ av)
{
    __shared__ float sc[16 * 2048];
    __shared__ float red[16][8];
    __shared__ float rowstat[16];

    const int i0 = blockIdx.x * 16;
    const int bt = blockIdx.y >> 4;     // batch
    const int h  = blockIdx.y & 15;     // head
    const int tid  = threadIdx.x;
    const int lane = tid & 31;
    const int wave = tid >> 5;          // 4 waves
    const int ml = lane & 15;
    const int hf = lane >> 4;

    const int ct_hi = (i0 >> 4) + 65;                    // AC tiles needed (<=128)
    const int ct_lo = (i0 >= 1008) ? 0 : ((1008 - i0) >> 4);  // first BD tile
    const int kmax  = ((i0 + 1040 + 31) >> 5) << 5;      // PV K extent (mult of 32)

    // ---- A fragments: (q + r_w_bias) and (q + r_r_bias), two K=32 chunks
    BF16Frag a_ac[2], a_bd[2];
    {
        const size_t qbase = (size_t)((DIM_MLEN + i0 + ml) * 2 + bt) * QKV_N + h * DIM_DH;
        #pragma unroll
        for (int c = 0; c < 2; ++c) {
            #pragma unroll
            for (int p = 0; p < 8; ++p) {
                int k = 2 * (p & 3) + 16 * (p >> 2) + 8 * hf + 32 * c;
                float q0 = bf2f(wh[qbase + k]);
                float q1 = bf2f(wh[qbase + k + 1]);
                a_ac[c].e[2 * p]     = f2bf(q0 + rwb[h * DIM_DH + k]);
                a_ac[c].e[2 * p + 1] = f2bf(q1 + rwb[h * DIM_DH + k + 1]);
                a_bd[c].e[2 * p]     = f2bf(q0 + rrb[h * DIM_DH + k]);
                a_bd[c].e[2 * p + 1] = f2bf(q1 + rrb[h * DIM_DH + k + 1]);
            }
        }
    }

    // ---- Phase 1: AC strip
    for (int ct = wave; ct < ct_hi; ct += 4) {
        int j0 = ct * 16;
        size_t kbase = (size_t)((j0 + ml) * 2 + bt) * QKV_N + DIM_D + h * DIM_DH;
        BF16Frag b0, b1;
        #pragma unroll
        for (int p = 0; p < 8; ++p) {
            int kk = 2 * (p & 3) + 16 * (p >> 2) + 8 * hf;
            b0.u[p] = *(const unsigned*)(wh + kbase + kk);
            b1.u[p] = *(const unsigned*)(wh + kbase + 32 + kk);
        }
        v8f acc = {};
        acc = wmma_bf16(a_ac[0], b0, acc);
        acc = wmma_bf16(a_ac[1], b1, acc);
        #pragma unroll
        for (int v = 0; v < 8; ++v)
            sc[(v + 8 * hf) * 2048 + j0 + ml] = acc[v];
    }
    __syncthreads();

    // ---- Phase 2: BD strip, scatter-add with rel-shift j = jj - 1023 + i
    for (int ct = ct_lo + wave; ct < 128; ct += 4) {
        int jj0 = ct * 16;
        size_t rbase = (size_t)(jj0 + ml) * DIM_D + h * DIM_DH;
        BF16Frag b0, b1;
        #pragma unroll
        for (int p = 0; p < 8; ++p) {
            int kk = 2 * (p & 3) + 16 * (p >> 2) + 8 * hf;
            b0.u[p] = *(const unsigned*)(rk + rbase + kk);
            b1.u[p] = *(const unsigned*)(rk + rbase + 32 + kk);
        }
        v8f acc = {};
        acc = wmma_bf16(a_bd[0], b0, acc);
        acc = wmma_bf16(a_bd[1], b1, acc);
        #pragma unroll
        for (int v = 0; v < 8; ++v) {
            int m = v + 8 * hf;
            int j = jj0 + ml - 1023 + i0 + m;
            if (j >= 0 && j < 2048)                 // wrap region == masked region
                sc[m * 2048 + j] += acc[v];
        }
    }
    __syncthreads();

    // ---- Phase 3: masked softmax over j (valid: j <= 1024 + i), scale=1/8
    {
        const float scale = 0.125f;
        int m = tid >> 3;         // row 0..15
        int sub = tid & 7;        // 8 threads per row
        int jmax = 1025 + i0 + m; // exclusive, <= 2048
        float lmax = -3.0e38f;
        for (int j = sub; j < jmax; j += 8)
            lmax = fmaxf(lmax, sc[m * 2048 + j] * scale);
        red[m][sub] = lmax;
        __syncthreads();
        if (sub == 0) {
            float r = red[m][0];
            #pragma unroll
            for (int t = 1; t < 8; ++t) r = fmaxf(r, red[m][t]);
            rowstat[m] = r;
        }
        __syncthreads();
        float rmax = rowstat[m];
        float lsum = 0.f;
        for (int j = sub; j < kmax; j += 8) {
            float e = (j < jmax) ? __expf(sc[m * 2048 + j] * scale - rmax) : 0.f;
            sc[m * 2048 + j] = e;
            lsum += e;
        }
        red[m][sub] = lsum;
        __syncthreads();
        if (sub == 0) {
            float r = 0.f;
            #pragma unroll
            for (int t = 0; t < 8; ++t) r += red[m][t];
            rowstat[m] = 1.0f / r;
        }
        __syncthreads();
        float inv = rowstat[m];
        for (int j = sub; j < jmax; j += 8)
            sc[m * 2048 + j] *= inv;
    }
    __syncthreads();

    // ---- Phase 4: out(16x64) = prob(16x2048) @ V(2048x64); wave -> 16-col tile
    {
        int nt = wave * 16;
        v8f acc = {};
        for (int kb = 0; kb < kmax; kb += 32) {
            BF16Frag a, bv;
            #pragma unroll
            for (int p = 0; p < 8; ++p) {
                int k = 2 * (p & 3) + 16 * (p >> 2) + 8 * hf + kb;
                a.e[2 * p]     = f2bf(sc[ml * 2048 + k]);
                a.e[2 * p + 1] = f2bf(sc[ml * 2048 + k + 1]);
            }
            #pragma unroll
            for (int p = 0; p < 8; ++p) {
                int kk = 2 * (p & 3) + 16 * (p >> 2) + 8 * hf;
                size_t r0 = (size_t)((kb + kk) * 2 + bt) * QKV_N + 2 * DIM_D + h * DIM_DH + nt + ml;
                bv.e[2 * p]     = wh[r0];
                bv.e[2 * p + 1] = wh[r0 + 2 * QKV_N];
            }
            acc = wmma_bf16(a, bv, acc);
        }
        #pragma unroll
        for (int v = 0; v < 8; ++v) {
            int row = (i0 + v + 8 * hf) * 2 + bt;
            int col = h * DIM_DH + nt + ml;
            av[(size_t)row * DIM_D + col] = f2bf(acc[v]);
        }
    }
}

// ---------------------------------------------------------------------------
// LayerNorm over D=1024 of (x + y), writes f32 (+ optional bf16 and f32 copy)
// ---------------------------------------------------------------------------
__global__ __launch_bounds__(256) void k_layernorm(const float* __restrict__ x,
                                                   const float* __restrict__ y,
                                                   const float* __restrict__ g,
                                                   const float* __restrict__ bta,
                                                   float* __restrict__ out,
                                                   __bf16* __restrict__ out_bf,
                                                   float* __restrict__ out_copy)
{
    __shared__ float r1[256], r2[256];
    const size_t base = (size_t)blockIdx.x * DIM_D;
    const int tid = threadIdx.x;
    float v[4];
    float s = 0.f, sq = 0.f;
    #pragma unroll
    for (int i = 0; i < 4; ++i) {
        int c = tid + i * 256;
        float t = x[base + c] + y[base + c];
        v[i] = t; s += t; sq += t * t;
    }
    r1[tid] = s; r2[tid] = sq;
    __syncthreads();
    for (int off = 128; off > 0; off >>= 1) {
        if (tid < off) { r1[tid] += r1[tid + off]; r2[tid] += r2[tid + off]; }
        __syncthreads();
    }
    float mean = r1[0] * (1.0f / 1024.0f);
    float var  = r2[0] * (1.0f / 1024.0f) - mean * mean;
    float rstd = rsqrtf(var + 1e-5f);
    #pragma unroll
    for (int i = 0; i < 4; ++i) {
        int c = tid + i * 256;
        float o = (v[i] - mean) * rstd * g[c] + bta[c];
        out[base + c] = o;
        if (out_bf)   out_bf[base + c] = f2bf(o);
        if (out_copy) out_copy[base + c] = o;
    }
}

// ---------------------------------------------------------------------------
// host orchestration
// ---------------------------------------------------------------------------
static inline dim3 g1(long n, int b) { return dim3((unsigned)((n + b - 1) / b)); }

extern "C" void kernel_launch(void* const* d_in, const int* in_sizes, int n_in,
                              void* d_out, int out_size, void* d_ws, size_t ws_size,
                              hipStream_t stream)
{
    (void)in_sizes; (void)n_in; (void)out_size; (void)ws_size;
    const float* mems = (const float*)d_in[0];
    const float* raw  = (const float*)d_in[1];
    // d_in[2] attention_mask is reproduced analytically (j > 1024 + i)
    const float* rwb  = (const float*)d_in[3];
    const float* rrb  = (const float*)d_in[4];
    const float* qkvw = (const float*)d_in[5];
    const float* rww  = (const float*)d_in[6];
    const float* oww  = (const float*)d_in[7];
    const float* ln1g = (const float*)d_in[8];
    const float* ln1b = (const float*)d_in[9];
    const float* ffw1 = (const float*)d_in[10];
    const float* ffb1 = (const float*)d_in[11];
    const float* ffw2 = (const float*)d_in[12];
    const float* ffb2 = (const float*)d_in[13];
    const float* ln2g = (const float*)d_in[14];
    const float* ln2b = (const float*)d_in[15];

    float* out = (float*)d_out;           // [0, CORE) = final core
    float* nm  = out + CORE_ELEMS;        // new_mems: 5 slots of CORE

    // --- workspace carve ---
    char* wp = (char*)d_ws;
    auto carve = [&](size_t bytes) { char* r = wp; wp += (bytes + 255) & ~(size_t)255; return r; };
    __bf16* wq  = (__bf16*)carve((size_t)DIM_NL * DIM_D * QKV_N * 2);   // qkv weights bf16
    __bf16* wr  = (__bf16*)carve((size_t)DIM_NL * DIM_D * DIM_D * 2);   // r_w
    __bf16* wo  = (__bf16*)carve((size_t)DIM_NL * DIM_D * DIM_D * 2);   // o_w
    __bf16* w1  = (__bf16*)carve((size_t)DIM_NL * DIM_D * DIM_D * 2);   // ff_w1
    __bf16* w2  = (__bf16*)carve((size_t)DIM_NL * DIM_D * DIM_D * 2);   // ff_w2
    __bf16* pe  = (__bf16*)carve((size_t)DIM_KLEN * DIM_D * 2);         // pos_emb
    __bf16* rk  = (__bf16*)carve((size_t)DIM_KLEN * DIM_D * 2);         // r_head_k (per layer)
    __bf16* cat = (__bf16*)carve((size_t)ROWS_CAT * DIM_D * 2);         // concat(mems,core) bf16
    __bf16* wh  = (__bf16*)carve((size_t)ROWS_CAT * QKV_N * 2);         // w_heads bf16
    __bf16* av  = (__bf16*)carve((size_t)ROWS_Q * DIM_D * 2);           // attn_vec bf16
    __bf16* hbf = (__bf16*)carve((size_t)ROWS_Q * DIM_D * 2);           // ln1 out bf16
    __bf16* f1  = (__bf16*)carve((size_t)ROWS_Q * DIM_D * 2);           // ff1 out bf16
    float*  core = (float*)carve((size_t)ROWS_Q * DIM_D * 4);           // current core f32
    float*  tmp  = (float*)carve((size_t)ROWS_Q * DIM_D * 4);           // attn_out / ff2 out f32
    float*  hf32 = (float*)carve((size_t)ROWS_Q * DIM_D * 4);           // ln1 out f32

    // --- setup: weight conversion, pos_emb, core init, new_mems[0] ---
    k_f32_to_bf16<<<g1((long)DIM_NL * DIM_D * QKV_N, 256), 256, 0, stream>>>(qkvw, wq, (long)DIM_NL * DIM_D * QKV_N);
    k_f32_to_bf16<<<g1((long)DIM_NL * DIM_D * DIM_D, 256), 256, 0, stream>>>(rww, wr, (long)DIM_NL * DIM_D * DIM_D);
    k_f32_to_bf16<<<g1((long)DIM_NL * DIM_D * DIM_D, 256), 256, 0, stream>>>(oww, wo, (long)DIM_NL * DIM_D * DIM_D);
    k_f32_to_bf16<<<g1((long)DIM_NL * DIM_D * DIM_D, 256), 256, 0, stream>>>(ffw1, w1, (long)DIM_NL * DIM_D * DIM_D);
    k_f32_to_bf16<<<g1((long)DIM_NL * DIM_D * DIM_D, 256), 256, 0, stream>>>(ffw2, w2, (long)DIM_NL * DIM_D * DIM_D);
    k_pos_emb<<<g1((long)DIM_KLEN * DIM_D, 256), 256, 0, stream>>>(pe);
    k_copy_f32<<<g1(CORE_ELEMS, 256), 256, 0, stream>>>(raw, core, CORE_ELEMS);
    k_copy_f32<<<g1(CORE_ELEMS, 256), 256, 0, stream>>>(raw, nm, CORE_ELEMS);   // hids[0]

    for (int l = 0; l < DIM_NL; ++l) {
        const __bf16* wql = wq + (size_t)l * DIM_D * QKV_N;
        const __bf16* wrl = wr + (size_t)l * DIM_D * DIM_D;
        const __bf16* wol = wo + (size_t)l * DIM_D * DIM_D;
        const __bf16* w1l = w1 + (size_t)l * DIM_D * DIM_D;
        const __bf16* w2l = w2 + (size_t)l * DIM_D * DIM_D;

        // cat = [mems[l]; core] -> bf16
        k_concat_bf16<<<g1((long)ROWS_CAT * DIM_D, 256), 256, 0, stream>>>(
            mems + (size_t)l * CORE_ELEMS, core, cat, (long)ROWS_CAT * DIM_D);

        // w_heads = cat @ qkv_w[l]   (4096 x 3072, K=1024) -> bf16
        k_gemm<1><<<dim3(QKV_N / 128, ROWS_CAT / 128), 256, 0, stream>>>(cat, wql, wh, nullptr, QKV_N, DIM_D);
        // r_head_k = pos_emb @ r_w[l] (2048 x 1024) -> bf16
        k_gemm<1><<<dim3(DIM_D / 128, DIM_KLEN / 128), 256, 0, stream>>>(pe, wrl, rk, nullptr, DIM_D, DIM_D);

        // attention: grid (64 query tiles, B*NH)
        k_attention<<<dim3(DIM_L / 16, DIM_B * DIM_NH), 128, 0, stream>>>(wh, rk, rwb, rrb, av);

        // attn_out = attn_vec @ o_w[l] (2048 x 1024) -> f32
        k_gemm<0><<<dim3(DIM_D / 128, ROWS_Q / 128), 256, 0, stream>>>(av, wol, tmp, nullptr, DIM_D, DIM_D);

        // h = LN(core + attn_out) -> f32 + bf16
        k_layernorm<<<ROWS_Q, 256, 0, stream>>>(core, tmp, ln1g + l * DIM_D, ln1b + l * DIM_D,
                                                hf32, hbf, nullptr);

        // f1 = relu(h @ ff_w1 + b1) -> bf16
        k_gemm<2><<<dim3(DIM_D / 128, ROWS_Q / 128), 256, 0, stream>>>(hbf, w1l, f1, ffb1 + l * DIM_D, DIM_D, DIM_D);
        // f2 = f1 @ ff_w2 + b2 -> f32
        k_gemm<3><<<dim3(DIM_D / 128, ROWS_Q / 128), 256, 0, stream>>>(f1, w2l, tmp, ffb2 + l * DIM_D, DIM_D, DIM_D);

        // core = LN(h + f2); also write new_mems[l+1]
        k_layernorm<<<ROWS_Q, 256, 0, stream>>>(hf32, tmp, ln2g + l * DIM_D, ln2b + l * DIM_D,
                                                core, nullptr, nm + (size_t)(l + 1) * CORE_ELEMS);
    }

    // final core -> d_out[0:CORE)
    k_copy_f32<<<g1(CORE_ELEMS, 256), 256, 0, stream>>>(core, out, CORE_ELEMS);
}
